// SE3TransformerIt_49624052138202
// MI455X (gfx1250) — matrix operand order repaired and checked by reading, not compiled, not run on hardware
//
#include <hip/hip_runtime.h>
#include <hip/hip_bf16.h>
#include <cstdint>

// ---------------------------------------------------------------------------
// CDNA5 (gfx1250) wave32 WMMA bf16 implementation of the Evoformer-ish block.
// All matmuls run on v_wmma_f32_16x16x32_bf16; activations kept in bf16 to
// minimize HBM traffic (workload is memory-bound on MI455X: ~1 TFLOP vs
// ~3-4 GB of traffic @ 23.3 TB/s).  Global tile loads are 16B-vectorized and
// the next k-chunk is prefetched (global_prefetch_b8) under the WMMA burst.
// ---------------------------------------------------------------------------

typedef __attribute__((ext_vector_type(16))) __bf16 bf16x16;
typedef __attribute__((ext_vector_type(8)))  __bf16 bf16x8;
typedef __attribute__((ext_vector_type(8)))  float  f32x8;

#define DEV __device__ __forceinline__

DEV unsigned short f2bf(float f) {
  unsigned int u = __float_as_uint(f);
  return (unsigned short)((u + 0x7fffu + ((u >> 16) & 1u)) >> 16);
}
DEV float geluf(float v) { return 0.5f * v * (1.0f + erff(v * 0.70710678118654752440f)); }

// A-fragment (16x32 bf16, MxK): lanes 0-15 hold K 0-7 (elems 0-7) and 16-23
// (elems 8-15); lanes 16-31 hold K 8-15 and 24-31.  Source is a row-major
// [m][k] LDS row pointer (k-chunk base already added).
DEV bf16x16 load_a_frag(const unsigned short* p, int hf) {
  const bf16x8 lo = *(const bf16x8*)(p + hf * 8);
  const bf16x8 hi = *(const bf16x8*)(p + 16 + hf * 8);
  return __builtin_shufflevector(lo, hi, 0,1,2,3,4,5,6,7,8,9,10,11,12,13,14,15);
}
// B-fragment (32x16 bf16, KxN): lanes 0-15 hold K 0-7 and 16-23 for col n=lane,
// lanes 16-31 hold K 8-15 and 24-31.  Source is column-major-in-LDS [n][k] row pointer.
DEV bf16x16 load_b_frag(const unsigned short* p, int hf) {
  const bf16x8 lo = *(const bf16x8*)(p + hf * 8);
  const bf16x8 hi = *(const bf16x8*)(p + 16 + hf * 8);
  return __builtin_shufflevector(lo, hi, 0,1,2,3,4,5,6,7,8,9,10,11,12,13,14,15);
}
DEV f32x8 wmma_bf16(bf16x16 a, bf16x16 b, f32x8 c) {
  return __builtin_amdgcn_wmma_f32_16x16x32_bf16(false, a, false, b, (short)0, c,
                                                 false, false);
}

// ---------------------------------------------------------------------------
// f32 -> bf16 converter (with scale), used for weights and the shared att map.
// ---------------------------------------------------------------------------
__global__ void cvt_bf16_kernel(const float* __restrict__ in,
                                unsigned short* __restrict__ out,
                                float scale, int n) {
  int i = blockIdx.x * 256 + threadIdx.x;
  if (i < n) out[i] = f2bf(in[i] * scale);
}

// ---------------------------------------------------------------------------
// LayerNorm over C=512, one row per 256-thread block, bf16 output.
// ---------------------------------------------------------------------------
__global__ __launch_bounds__(256) void ln_kernel(const float* __restrict__ x,
                                                 const float* __restrict__ g,
                                                 const float* __restrict__ bta,
                                                 unsigned short* __restrict__ out) {
  const size_t base = (size_t)blockIdx.x * 512;
  const int tid = threadIdx.x;
  float v0 = x[base + tid], v1 = x[base + tid + 256];
  float s = v0 + v1;
#pragma unroll
  for (int off = 16; off >= 1; off >>= 1) s += __shfl_xor(s, off, 32);
  __shared__ float red[8], red2[8];
  int w = tid >> 5;
  if ((tid & 31) == 0) red[w] = s;
  __syncthreads();
  float tot = red[0]+red[1]+red[2]+red[3]+red[4]+red[5]+red[6]+red[7];
  float mu = tot * (1.0f / 512.0f);
  float d0 = v0 - mu, d1 = v1 - mu;
  float q = d0 * d0 + d1 * d1;
#pragma unroll
  for (int off = 16; off >= 1; off >>= 1) q += __shfl_xor(q, off, 32);
  if ((tid & 31) == 0) red2[w] = q;
  __syncthreads();
  float var = (red2[0]+red2[1]+red2[2]+red2[3]+red2[4]+red2[5]+red2[6]+red2[7]) *
              (1.0f / 512.0f);
  float rstd = rsqrtf(var + 1e-5f);
  out[base + tid]       = f2bf(d0 * rstd * g[tid]       + bta[tid]);
  out[base + tid + 256] = f2bf(d1 * rstd * g[tid + 256] + bta[tid + 256]);
}

// ---------------------------------------------------------------------------
// Tiled bf16 WMMA GEMM:  out[z][r][c] = epilogue( A[z][r][:] @ B[z][:][c] ).
// Macro tile 128x128, 256 threads = 8 waves in a 2x4 grid of 64x32 wave tiles.
// MODE 0: bf16 store, (acc+bias)*scale            (projections / batched attn)
// MODE 1: f32 store, acc+bias+res; optional out2 slice at m==0 (col proj)
// MODE 2: bf16 store, gelu(acc+bias)              (MLP mid)
// ---------------------------------------------------------------------------
template <int MODE>
__global__ __launch_bounds__(256) void gemm_kernel(
    const unsigned short* __restrict__ A, const unsigned short* __restrict__ B,
    const float* __restrict__ bias, const float* __restrict__ res,
    void* __restrict__ out, float* __restrict__ out2,
    int K, int ncols,
    long long aStrideZ, int aZdiv, long long bStrideZ, long long oStrideZ,
    float scale) {
  __shared__ alignas(64) unsigned short sA[128][32];
  __shared__ alignas(64) unsigned short sB[128][32];  // [n][k]

  const int tid = threadIdx.x;
  const int lane = tid & 31, hf = lane >> 4, l16 = lane & 15;
  const int wid = tid >> 5;
  const int wm = wid >> 2;   // 0..1
  const int wn = wid & 3;    // 0..3
  const int rowBase = blockIdx.y * 128;
  const int colBase = blockIdx.x * 128;
  const int z = blockIdx.z;

  const unsigned short* Az = A + (long long)(z / aZdiv) * aStrideZ + (size_t)rowBase * K;
  const unsigned short* Bz = B + (long long)z * bStrideZ + colBase;

  const f32x8 vzero = {0.f,0.f,0.f,0.f,0.f,0.f,0.f,0.f};
  f32x8 acc[4][2];
#pragma unroll
  for (int mi = 0; mi < 4; ++mi)
#pragma unroll
    for (int ni = 0; ni < 2; ++ni) acc[mi][ni] = vzero;

  for (int k0 = 0; k0 < K; k0 += 32) {
    // A tile: 128x32 bf16, 16B vector copies (2 per thread).
    for (int i = tid; i < 512; i += 256) {
      int rr = i >> 2, kk = (i & 3) << 3;
      *(uint4*)(&sA[rr][kk]) = *(const uint4*)(Az + (size_t)rr * K + k0 + kk);
      // Prefetch next k-chunk of A (speculative; silently dropped if OOB).
      __builtin_prefetch(Az + (size_t)rr * K + k0 + 32 + kk, 0, 1);
    }
    // B tile: 32x128 bf16, 16B vector loads, transposed scatter into [n][k].
    for (int i = tid; i < 512; i += 256) {
      int kk = i >> 4;            // 0..31
      int nn = (i & 15) << 3;     // 0,8,...,120
      uint4 pv = *(const uint4*)(Bz + (size_t)(k0 + kk) * ncols + nn);
      __builtin_prefetch(Bz + (size_t)(k0 + 32 + kk) * ncols + nn, 0, 1);
      const unsigned short* pe = (const unsigned short*)&pv;
#pragma unroll
      for (int j = 0; j < 8; ++j) sB[nn + j][kk] = pe[j];
    }
    __syncthreads();

    bf16x16 af[4], bv[2];
#pragma unroll
    for (int mi = 0; mi < 4; ++mi)
      af[mi] = load_a_frag(&sA[wm * 64 + mi * 16 + l16][0], hf);
#pragma unroll
    for (int ni = 0; ni < 2; ++ni)
      bv[ni] = load_b_frag(&sB[wn * 32 + ni * 16 + l16][0], hf);
#pragma unroll
    for (int mi = 0; mi < 4; ++mi)
#pragma unroll
      for (int ni = 0; ni < 2; ++ni)
        acc[mi][ni] = wmma_bf16(af[mi], bv[ni], acc[mi][ni]);
    __syncthreads();
  }

#pragma unroll
  for (int mi = 0; mi < 4; ++mi)
#pragma unroll
    for (int ni = 0; ni < 2; ++ni)
#pragma unroll
      for (int r = 0; r < 8; ++r) {
        int row = rowBase + wm * 64 + mi * 16 + r + 8 * hf;
        int col = colBase + wn * 32 + ni * 16 + l16;
        size_t oidx = (size_t)z * oStrideZ + (size_t)row * ncols + col;
        float v = acc[mi][ni][r];
        float bb = bias ? bias[col] : 0.0f;
        if (MODE == 0) {
          ((unsigned short*)out)[oidx] = f2bf((v + bb) * scale);
        } else if (MODE == 1) {
          float rv = res ? res[oidx] : 0.0f;
          ((float*)out)[oidx] = v + bb + rv;
          if (out2) {  // flat layout: row = ((b*128)+m)*256 + n
            int m = (row >> 8) & 127;
            if (m == 0) {
              int b = row >> 15, n = row & 255;
              out2[(size_t)((b << 8) + n) * 512 + col] = v + bb;
            }
          }
        } else {
          ((unsigned short*)out)[oidx] = f2bf(geluf(v + bb));
        }
      }
}

// ---------------------------------------------------------------------------
// Fused column attention: one block per (b, n, h); M=128 rows, D=64 head dim.
// S = (Q/8)K^T  -> softmax (register, shfl reductions) -> Y = S V.
// 64 KB LDS; V^T aliases Q's region after phase 1.
// ---------------------------------------------------------------------------
__global__ __launch_bounds__(128) void col_attn_kernel(
    const unsigned short* __restrict__ q, const unsigned short* __restrict__ k,
    const unsigned short* __restrict__ v, unsigned short* __restrict__ y) {
  __shared__ alignas(64) unsigned short smem[32768];
  unsigned short* sQ  = smem;           // [128][64]
  unsigned short* sK  = smem + 8192;    // [128][64]
  unsigned short* sS  = smem + 16384;   // [128][128]
  unsigned short* sVt = smem;           // [64][128], aliases sQ

  const int h = blockIdx.x, n = blockIdx.y, b = blockIdx.z;
  const int tid = threadIdx.x;
  const int lane = tid & 31, hf = lane >> 4, l16 = lane & 15;
  const int wid = tid >> 5;
  const int wr = wid * 32;
  const size_t NC = 256 * 512;
  const size_t base = ((size_t)(b * 128) * 256 + n) * 512 + (size_t)h * 64;

  for (int i = tid; i < 1024; i += 128) {
    int m = i >> 3, dc = (i & 7) << 3;
    *(uint4*)(sQ + m * 64 + dc) = *(const uint4*)(q + base + (size_t)m * NC + dc);
    *(uint4*)(sK + m * 64 + dc) = *(const uint4*)(k + base + (size_t)m * NC + dc);
    __builtin_prefetch(v + base + (size_t)m * NC + dc, 0, 1);  // warm V for phase 2
  }
  __syncthreads();

  const f32x8 vzero = {0.f,0.f,0.f,0.f,0.f,0.f,0.f,0.f};
  f32x8 sacc[2][8];
#pragma unroll
  for (int mi = 0; mi < 2; ++mi)
#pragma unroll
    for (int nj = 0; nj < 8; ++nj) sacc[mi][nj] = vzero;

#pragma unroll
  for (int kd = 0; kd < 64; kd += 32) {
    bf16x16 af[2];
#pragma unroll
    for (int mi = 0; mi < 2; ++mi)
      af[mi] = load_a_frag(sQ + (size_t)(wr + mi * 16 + l16) * 64 + kd, hf);
#pragma unroll
    for (int nj = 0; nj < 8; ++nj) {
      bf16x16 bv = load_b_frag(sK + (size_t)(nj * 16 + l16) * 64 + kd, hf);
#pragma unroll
      for (int mi = 0; mi < 2; ++mi) sacc[mi][nj] = wmma_bf16(af[mi], bv, sacc[mi][nj]);
    }
  }

  // Softmax over the 128 columns of each row this wave owns.  C-fragment rows
  // live in 16-lane groups (same half), so xor-8/4/2/1 butterflies reduce rows.
#pragma unroll
  for (int mi = 0; mi < 2; ++mi)
#pragma unroll
    for (int r = 0; r < 8; ++r) {
      int mrow = wr + mi * 16 + r + 8 * hf;
      float mx = -1e30f;
#pragma unroll
      for (int nj = 0; nj < 8; ++nj) mx = fmaxf(mx, sacc[mi][nj][r]);
#pragma unroll
      for (int off = 8; off >= 1; off >>= 1) mx = fmaxf(mx, __shfl_xor(mx, off, 32));
      float e[8], sum = 0.f;
#pragma unroll
      for (int nj = 0; nj < 8; ++nj) { e[nj] = __expf(sacc[mi][nj][r] - mx); sum += e[nj]; }
#pragma unroll
      for (int off = 8; off >= 1; off >>= 1) sum += __shfl_xor(sum, off, 32);
      float inv = 1.0f / sum;
#pragma unroll
      for (int nj = 0; nj < 8; ++nj)
        sS[(size_t)mrow * 128 + nj * 16 + l16] = f2bf(e[nj] * inv);
    }
  __syncthreads();

  // V^T into (former) Q region: 16B vector loads, transposed scatter.
  for (int i = tid; i < 1024; i += 128) {
    int m = i >> 3, dc = (i & 7) << 3;
    uint4 pv = *(const uint4*)(v + base + (size_t)m * NC + dc);
    const unsigned short* pe = (const unsigned short*)&pv;
#pragma unroll
    for (int j = 0; j < 8; ++j) sVt[(size_t)(dc + j) * 128 + m] = pe[j];
  }
  __syncthreads();

  f32x8 yacc[2][4];
#pragma unroll
  for (int mi = 0; mi < 2; ++mi)
#pragma unroll
    for (int nj = 0; nj < 4; ++nj) yacc[mi][nj] = vzero;

#pragma unroll
  for (int mk = 0; mk < 128; mk += 32) {
    bf16x16 af[2];
#pragma unroll
    for (int mi = 0; mi < 2; ++mi)
      af[mi] = load_a_frag(sS + (size_t)(wr + mi * 16 + l16) * 128 + mk, hf);
#pragma unroll
    for (int nj = 0; nj < 4; ++nj) {
      bf16x16 bv = load_b_frag(sVt + (size_t)(nj * 16 + l16) * 128 + mk, hf);
#pragma unroll
      for (int mi = 0; mi < 2; ++mi) yacc[mi][nj] = wmma_bf16(af[mi], bv, yacc[mi][nj]);
    }
  }

#pragma unroll
  for (int mi = 0; mi < 2; ++mi)
#pragma unroll
    for (int nj = 0; nj < 4; ++nj)
#pragma unroll
      for (int r = 0; r < 8; ++r) {
        int m = wr + mi * 16 + r + 8 * hf;
        int d = nj * 16 + l16;
        y[base + (size_t)m * NC + d] = f2bf(yacc[mi][nj][r]);
      }
}

// ---------------------------------------------------------------------------
// Host-side orchestration.
// ---------------------------------------------------------------------------
extern "C" void kernel_launch(void* const* d_in, const int* in_sizes, int n_in,
                              void* d_out, int out_size, void* d_ws, size_t ws_size,
                              hipStream_t stream) {
  (void)in_sizes; (void)n_in; (void)out_size; (void)ws_size;

  const float* x      = (const float*)d_in[0];
  const float* att    = (const float*)d_in[1];
  const float* ln_g   = (const float*)d_in[2];
  const float* ln_b   = (const float*)d_in[3];
  const float* col_wk = (const float*)d_in[4];  const float* col_bk = (const float*)d_in[5];
  const float* col_wq = (const float*)d_in[6];  const float* col_bq = (const float*)d_in[7];
  const float* col_wv = (const float*)d_in[8];  const float* col_bv = (const float*)d_in[9];
  const float* col_wp = (const float*)d_in[10]; const float* col_bp = (const float*)d_in[11];
  const float* row_wv = (const float*)d_in[12]; const float* row_bv = (const float*)d_in[13];
  const float* row_wp = (const float*)d_in[14]; const float* row_bp = (const float*)d_in[15];
  const float* mlp_w1 = (const float*)d_in[16]; const float* mlp_b1 = (const float*)d_in[17];
  const float* mlp_w2 = (const float*)d_in[18]; const float* mlp_b2 = (const float*)d_in[19];

  const size_t RC = 67108864ull;  // R*C = 131072*512
  char* ws = (char*)d_ws;
  // Workspace layout (~950 MB total):
  unsigned short* bufLN = (unsigned short*)(ws + 0);             // bf16 RC
  unsigned short* bufQ  = (unsigned short*)(ws + RC * 2);        // bf16 RC
  unsigned short* bufK  = (unsigned short*)(ws + RC * 4);        // bf16 RC
  unsigned short* bufV  = (unsigned short*)(ws + RC * 6);        // bf16 RC
  float*          bufX  = (float*)        (ws + RC * 8);         // f32  RC
  unsigned short* bufH  = (unsigned short*)(ws + RC * 8 + RC * 4);        // bf16 32768*2048
  unsigned short* wW    = (unsigned short*)(ws + RC * 8 + RC * 4 + 134217728ull);
  unsigned short* wk  = wW;            unsigned short* wq  = wW + 262144;
  unsigned short* wv  = wW + 524288;   unsigned short* wp  = wW + 786432;
  unsigned short* rwv = wW + 1048576;  unsigned short* rwp = wW + 1310720;
  unsigned short* w1  = wW + 1572864;  unsigned short* w2  = wW + 2621440;
  unsigned short* attb = wW + 3670016;

  float* outX = (float*)d_out;
  float* out2 = outX + RC;  // y[:,0,:,:] region, B*N*C floats

  auto CV = [&](const float* s, unsigned short* d, int n, float sc) {
    cvt_bf16_kernel<<<(n + 255) / 256, 256, 0, stream>>>(s, d, sc, n);
  };
  CV(col_wk, wk, 262144, 1.f);  CV(col_wq, wq, 262144, 1.f);
  CV(col_wv, wv, 262144, 1.f);  CV(col_wp, wp, 262144, 1.f);
  CV(row_wv, rwv, 262144, 1.f); CV(row_wp, rwp, 262144, 1.f);
  CV(mlp_w1, w1, 1048576, 1.f); CV(mlp_w2, w2, 1048576, 1.f);
  CV(att, attb, 262144, 0.125f);  // row-attn 1/sqrt(D) folded in

  dim3 blk(256);
  dim3 gFlat(4, 1024, 1);  // 512 cols x 131072 rows

  // ---- col attention branch ----
  ln_kernel<<<131072, 256, 0, stream>>>(x, ln_g, ln_b, bufLN);
  gemm_kernel<0><<<gFlat, blk, 0, stream>>>(bufLN, wq, col_bq, nullptr, bufQ, nullptr,
                                            512, 512, 0, 1, 0, 0, 0.125f);
  gemm_kernel<0><<<gFlat, blk, 0, stream>>>(bufLN, wk, col_bk, nullptr, bufK, nullptr,
                                            512, 512, 0, 1, 0, 0, 1.f);
  gemm_kernel<0><<<gFlat, blk, 0, stream>>>(bufLN, wv, col_bv, nullptr, bufV, nullptr,
                                            512, 512, 0, 1, 0, 0, 1.f);
  col_attn_kernel<<<dim3(8, 256, 4), 128, 0, stream>>>(bufQ, bufK, bufV, bufLN);
  gemm_kernel<1><<<gFlat, blk, 0, stream>>>(bufLN, wp, col_bp, x, bufX, out2,
                                            512, 512, 0, 1, 0, 0, 1.f);

  // ---- row attention branch ----
  ln_kernel<<<131072, 256, 0, stream>>>(bufX, ln_g, ln_b, bufQ);
  gemm_kernel<0><<<gFlat, blk, 0, stream>>>(bufQ, rwv, row_bv, nullptr, bufK, nullptr,
                                            512, 512, 0, 1, 0, 0, 1.f);
  // yrow[b,m] = (att[b]/8) @ vrow[b,m]  :  z in [0, B*M), A advances per b.
  gemm_kernel<0><<<dim3(4, 2, 512), blk, 0, stream>>>(attb, bufK, nullptr, nullptr,
                                                      bufV, nullptr, 256, 512,
                                                      65536, 128, 131072, 131072, 1.f);
  gemm_kernel<1><<<gFlat, blk, 0, stream>>>(bufV, rwp, row_bp, bufX, bufX, nullptr,
                                            512, 512, 0, 1, 0, 0, 1.f);

  // ---- MLP (4 row-chunks of 32768 to bound workspace) ----
  ln_kernel<<<131072, 256, 0, stream>>>(bufX, ln_g, ln_b, bufQ);
  for (int c = 0; c < 4; ++c) {
    size_t r0 = (size_t)c * 32768;
    gemm_kernel<2><<<dim3(16, 256, 1), blk, 0, stream>>>(bufQ + r0 * 512, w1, mlp_b1,
                                                         nullptr, bufH, nullptr,
                                                         512, 2048, 0, 1, 0, 0, 1.f);
    gemm_kernel<1><<<dim3(4, 256, 1), blk, 0, stream>>>(bufH, w2, mlp_b2,
                                                        bufX + r0 * 512, outX + r0 * 512,
                                                        nullptr, 2048, 512, 0, 1, 0, 0, 1.f);
  }
}
